// CustomFlashAttention_70171175682168
// MI455X (gfx1250) — compile-verified, matrix operand order checked by reading, not run on hardware
//
#include <hip/hip_runtime.h>

// Problem constants (fixed by the reference)
#define SEQ   2048
#define DIM   2048
#define NHEAD 16
#define HDIM  128
#define VSTRIDE 40    // padded LDS row stride for V^T / GEMM tiles: 80B rows, 16B aligned
#define KSTRIDE 136   // padded LDS row stride for K tile: 272B rows, start bank 4n -> conflict-free

// Scheduling fence: keep the batched LDS fragment loads from being re-sunk
// next to each WMMA (round-3 asm showed load->wait->wmma serialization).
#if defined(__has_builtin)
#  if __has_builtin(__builtin_amdgcn_sched_barrier)
#    define SCHED_BARRIER() __builtin_amdgcn_sched_barrier(0)
#  endif
#endif
#ifndef SCHED_BARRIER
#  define SCHED_BARRIER() ((void)0)
#endif

typedef __attribute__((ext_vector_type(8)))  __bf16 v8bf;
typedef __attribute__((ext_vector_type(16))) __bf16 v16bf;
typedef __attribute__((ext_vector_type(8)))  float  v8f;

static __device__ __forceinline__ __bf16 f2bf(float f) {
  unsigned u = __builtin_bit_cast(unsigned, f);
  u += 0x7FFFu + ((u >> 16) & 1u);              // round-to-nearest-even
  unsigned short h = (unsigned short)(u >> 16);
  return __builtin_bit_cast(__bf16, h);
}

// Load one 16-bit WMMA fragment for a source contiguous along K:
// lanes 0-15 take K-chunks [0..7] and [16..23]; lanes 16-31 take [8..15] and [24..31].
// Caller pre-offsets p by (laneHi * 8); chunk1 is always at +16 elements.
static __device__ __forceinline__ v16bf load_frag(const __bf16* p) {
  v8bf a = *(const v8bf*)(p);
  v8bf b = *(const v8bf*)(p + 16);
  return __builtin_shufflevector(a, b, 0,1,2,3,4,5,6,7,8,9,10,11,12,13,14,15);
}

static __device__ __forceinline__ v8f wmma_bf(v16bf a, v16bf b, v8f c) {
  return __builtin_amdgcn_wmma_f32_16x16x32_bf16(false, a, false, b, (short)0, c,
                                                 false, false);
}

// ---------------------------------------------------------------------------
// fp32 -> bf16 conversion (x and the four weight matrices)
// ---------------------------------------------------------------------------
__global__ __launch_bounds__(256)
void cvt_f32_to_bf16(const float* __restrict__ in, __bf16* __restrict__ out) {
  size_t i = ((size_t)blockIdx.x * blockDim.x + threadIdx.x) * 4;
  float4 v = *(const float4*)(in + i);
  out[i + 0] = f2bf(v.x);
  out[i + 1] = f2bf(v.y);
  out[i + 2] = f2bf(v.z);
  out[i + 3] = f2bf(v.w);
}

// ---------------------------------------------------------------------------
// C[m,n] = sum_k A[m,k] * W[n,k]   (A @ W^T; both operands K-contiguous).
// Block: 256 threads = 8 waves; block tile 128x128, k-step 32.
// Double-buffered LDS staging: each k-step the 128x32 A and W tiles are loaded
// once per block (16KB) and reused by all 8 waves; global loads for step k+1
// are issued before the LDS fragment reads of step k so they overlap WMMAs.
// ---------------------------------------------------------------------------
template <bool F32OUT>
__global__ __launch_bounds__(256)
void gemm_xwt(const __bf16* __restrict__ A, const __bf16* __restrict__ W,
              void* __restrict__ Cout) {
  __shared__ __bf16 la[2][128 * VSTRIDE];   // 2 x 10KB
  __shared__ __bf16 lw[2][128 * VSTRIDE];

  const int tid    = threadIdx.x;
  const int wid    = tid >> 5;
  const int lane   = tid & 31;
  const int laneN  = lane & 15;
  const int laneHi = lane >> 4;
  const int rowB = blockIdx.y * 128;
  const int colB = blockIdx.x * 128;
  const int rowW = (wid & 3) * 32;          // 4 waves along M (32 rows each)
  const int colW = (wid >> 2) * 64;         // 2 waves along N (64 cols each)

  // Staging: thread owns rows r0 and r0+64 (one 16B chunk each) of both tiles.
  const int r0 = tid >> 2;
  const int ko = (tid & 3) * 8;
  const __bf16* pa0 = A + (size_t)(rowB + r0)      * DIM + ko;
  const __bf16* pa1 = A + (size_t)(rowB + r0 + 64) * DIM + ko;
  const __bf16* pw0 = W + (size_t)(colB + r0)      * DIM + ko;
  const __bf16* pw1 = W + (size_t)(colB + r0 + 64) * DIM + ko;
  const int lo0 = r0 * VSTRIDE + ko;
  const int lo1 = (r0 + 64) * VSTRIDE + ko;

  v8bf ra0 = *(const v8bf*)(pa0);
  v8bf ra1 = *(const v8bf*)(pa1);
  v8bf rw0 = *(const v8bf*)(pw0);
  v8bf rw1 = *(const v8bf*)(pw1);
  *(v8bf*)&la[0][lo0] = ra0;
  *(v8bf*)&la[0][lo1] = ra1;
  *(v8bf*)&lw[0][lo0] = rw0;
  *(v8bf*)&lw[0][lo1] = rw1;
  __syncthreads();

  v8f acc[2][4] = {};

  for (int ks = 0; ks < DIM / 32; ++ks) {
    const int cur  = ks & 1;
    const bool more = (ks + 1) < (DIM / 32);
    if (more) {
      const int k0 = (ks + 1) * 32;
      ra0 = *(const v8bf*)(pa0 + k0);
      ra1 = *(const v8bf*)(pa1 + k0);
      rw0 = *(const v8bf*)(pw0 + k0);
      rw1 = *(const v8bf*)(pw1 + k0);
    }

    v16bf af[2], wf[4];
#pragma unroll
    for (int mi = 0; mi < 2; ++mi)
      af[mi] = load_frag(&la[cur][(rowW + mi * 16 + laneN) * VSTRIDE + laneHi * 8]);
#pragma unroll
    for (int ni = 0; ni < 4; ++ni)
      wf[ni] = load_frag(&lw[cur][(colW + ni * 16 + laneN) * VSTRIDE + laneHi * 8]);
    SCHED_BARRIER();   // keep the 12 ds_load_b128 batched ahead of the WMMAs
#pragma unroll
    for (int mi = 0; mi < 2; ++mi)
#pragma unroll
      for (int ni = 0; ni < 4; ++ni)
        acc[mi][ni] = wmma_bf(af[mi], wf[ni], acc[mi][ni]);

    if (more) {
      const int nxt = cur ^ 1;
      *(v8bf*)&la[nxt][lo0] = ra0;
      *(v8bf*)&la[nxt][lo1] = ra1;
      *(v8bf*)&lw[nxt][lo0] = rw0;
      *(v8bf*)&lw[nxt][lo1] = rw1;
    }
    __syncthreads();
  }

#pragma unroll
  for (int mi = 0; mi < 2; ++mi)
#pragma unroll
    for (int ni = 0; ni < 4; ++ni)
#pragma unroll
      for (int r = 0; r < 8; ++r) {
        const int row = rowB + rowW + mi * 16 + r + laneHi * 8;
        const int col = colB + colW + ni * 16 + laneN;
        const float v = acc[mi][ni][r];
        if (F32OUT)
          ((float*)Cout)[(size_t)row * DIM + col] = v;
        else
          ((__bf16*)Cout)[(size_t)row * DIM + col] = f2bf(v);
      }
}

// ---------------------------------------------------------------------------
// Flash attention.  Grid: 16 heads x 16 q-tiles (128 rows).  Block: 8 waves,
// each wave owns 16 query rows.  Scores computed transposed (keys x queries)
// so softmax is in-lane; the S^T f32 C-layout maps 1:1 onto the bf16
// A-fragment layout of P (no cross-lane movement).  K and V tiles staged in
// LDS once per block; V stored transposed so PV B-fragments are contiguous.
// sched_barrier(0) pins the batched fragment loads ahead of each WMMA burst.
// ---------------------------------------------------------------------------
__global__ __launch_bounds__(256)
void fa_kernel(const __bf16* __restrict__ Q, const __bf16* __restrict__ K,
               const __bf16* __restrict__ V, __bf16* __restrict__ O) {
  __shared__ __bf16 kT[32 * KSTRIDE];     // K tile  [key][d],  8.7 KB
  __shared__ __bf16 vT[HDIM * VSTRIDE];   // V^T tile [d][key], 10.2 KB

  const int head   = blockIdx.x & (NHEAD - 1);
  const int qt     = blockIdx.x >> 4;
  const int tid    = threadIdx.x;
  const int wid    = tid >> 5;
  const int lane   = tid & 31;
  const int laneN  = lane & 15;
  const int laneHi = lane >> 4;
  const size_t hoff = (size_t)head * HDIM;
  const int q0 = qt * 128 + wid * 16;

  // Q as B-fragments (d x queries), 4 chunks of K=32 along d, held for all keys.
  v16bf qf[4];
#pragma unroll
  for (int c = 0; c < 4; ++c)
    qf[c] = load_frag(Q + (size_t)(q0 + laneN) * DIM + hoff + c * 32 + laneHi * 8);

  v8f o[8] = {};                    // 16 queries x 128 d accumulators
  float m = -__builtin_inff();
  float l = 0.0f;
  const float scale = 0.08838834764831845f;   // 1/sqrt(128)
  const float L2E   = 1.44269504088896f;

  for (int kb = 0; kb < SEQ; kb += 32) {
    __syncthreads();  // previous iteration's kT/vT reads complete

    // ---- stage K tile [key][d] (row-major, contiguous 16B copies) ----
    for (int i = tid; i < 512; i += 256) {      // 512 chunks of 8 bf16
      const int key = i >> 4;
      const int dc  = (i & 15) << 3;
      *(v8bf*)&kT[key * KSTRIDE + dc] =
          *(const v8bf*)(K + (size_t)(kb + key) * DIM + hoff + dc);
    }
    // ---- stage V^T tile [d][key] (transposing scalar stores) ----
    for (int i = tid; i < 512; i += 256) {
      const int key = i >> 4;
      const int dc  = (i & 15) << 3;
      v8bf val = *(const v8bf*)(V + (size_t)(kb + key) * DIM + hoff + dc);
#pragma unroll
      for (int e = 0; e < 8; ++e) vT[(dc + e) * VSTRIDE + key] = val[e];
    }
    __syncthreads();

    // ---- S^T = K_tile (A: keys x d) @ Q^T (B: d x queries), two 16-key tiles.
    // All 8 K A-fragments (16 ds_load_b128) are issued as one batch, then the
    // 8 WMMAs run back-to-back (alternating accumulators -> no RAW stalls).
    v16bf ka[8];
#pragma unroll
    for (int c = 0; c < 4; ++c) {
      ka[c]     = load_frag(&kT[laneN        * KSTRIDE + c * 32 + laneHi * 8]);
      ka[4 + c] = load_frag(&kT[(16 + laneN) * KSTRIDE + c * 32 + laneHi * 8]);
    }
    SCHED_BARRIER();
    v8f c0 = {}, c1 = {};
#pragma unroll
    for (int c = 0; c < 4; ++c) {
      c0 = wmma_bf(ka[c],     qf[c], c0);
      c1 = wmma_bf(ka[4 + c], qf[c], c1);
    }

    // ---- online softmax (each lane: query laneN, 16 of 32 keys; partner lane+16)
    float s0[8], s1[8];
    float tmax = -__builtin_inff();
#pragma unroll
    for (int r = 0; r < 8; ++r) {
      s0[r] = c0[r] * scale;
      s1[r] = c1[r] * scale;
      tmax = fmaxf(tmax, fmaxf(s0[r], s1[r]));
    }
    tmax = fmaxf(tmax, __shfl_xor(tmax, 16, 32));
    const float mnew  = fmaxf(m, tmax);
    const float alpha = exp2f((m - mnew) * L2E);
    float rs = 0.0f;
    float p0[8], p1[8];
#pragma unroll
    for (int r = 0; r < 8; ++r) {
      p0[r] = exp2f((s0[r] - mnew) * L2E);
      p1[r] = exp2f((s1[r] - mnew) * L2E);
      rs += p0[r] + p1[r];
    }
    rs += __shfl_xor(rs, 16, 32);
    l = l * alpha + rs;
    m = mnew;

    // ---- rescale O accumulators (broadcast per-query alpha into C layout)
    float ab[8];
#pragma unroll
    for (int r = 0; r < 8; ++r) ab[r] = __shfl(alpha, r + laneHi * 8, 32);
#pragma unroll
    for (int t = 0; t < 8; ++t)
#pragma unroll
      for (int r = 0; r < 8; ++r) o[t][r] *= ab[r];

    // ---- pack P (16 queries x 32 keys) straight into A-fragment layout
    v16bf pf;
#pragma unroll
    for (int r = 0; r < 8; ++r) {
      pf[r]     = f2bf(p0[r]);   // keys [0..7]+8*laneHi
      pf[r + 8] = f2bf(p1[r]);   // keys [16..23]+8*laneHi
    }

    // ---- O += P @ V.  B-fragments loaded in two groups of 4 (8 ds_load_b128
    // per group) so loads batch without blowing up live VGPRs.
#pragma unroll
    for (int g = 0; g < 2; ++g) {
      v16bf vf[4];
#pragma unroll
      for (int t = 0; t < 4; ++t)
        vf[t] = load_frag(&vT[((g * 4 + t) * 16 + laneN) * VSTRIDE + laneHi * 8]);
      SCHED_BARRIER();
#pragma unroll
      for (int t = 0; t < 4; ++t)
        o[g * 4 + t] = wmma_bf(pf, vf[t], o[g * 4 + t]);
    }
  }

  // ---- normalize and store O (bf16) ----
  float lb[8];
#pragma unroll
  for (int r = 0; r < 8; ++r) lb[r] = __shfl(l, r + laneHi * 8, 32);
#pragma unroll
  for (int t = 0; t < 8; ++t)
#pragma unroll
    for (int r = 0; r < 8; ++r) {
      const int row = q0 + r + laneHi * 8;
      const int col = t * 16 + laneN;
      O[(size_t)row * DIM + hoff + col] = f2bf(o[t][r] / lb[r]);
    }
}

// ---------------------------------------------------------------------------
extern "C" void kernel_launch(void* const* d_in, const int* in_sizes, int n_in,
                              void* d_out, int out_size, void* d_ws, size_t ws_size,
                              hipStream_t stream) {
  (void)in_sizes; (void)n_in; (void)out_size; (void)ws_size;
  const float* x  = (const float*)d_in[0];
  const float* wq = (const float*)d_in[1];
  const float* wk = (const float*)d_in[2];
  const float* wv = (const float*)d_in[3];
  const float* wo = (const float*)d_in[4];
  float* out = (float*)d_out;

  const size_t nel = (size_t)SEQ * DIM;   // 4M elements per matrix
  __bf16* xb  = (__bf16*)d_ws;            // 9 x 8MB = 72MB workspace
  __bf16* wqb = xb  + nel;
  __bf16* wkb = wqb + nel;
  __bf16* wvb = wkb + nel;
  __bf16* wob = wvb + nel;
  __bf16* Qb  = wob + nel;
  __bf16* Kb  = Qb  + nel;
  __bf16* Vb  = Kb  + nel;
  __bf16* Ob  = Vb  + nel;

  const int cvtBlocks = (int)(nel / 4 / 256);   // 4096
  cvt_f32_to_bf16<<<cvtBlocks, 256, 0, stream>>>(x,  xb);
  cvt_f32_to_bf16<<<cvtBlocks, 256, 0, stream>>>(wq, wqb);
  cvt_f32_to_bf16<<<cvtBlocks, 256, 0, stream>>>(wk, wkb);
  cvt_f32_to_bf16<<<cvtBlocks, 256, 0, stream>>>(wv, wvb);
  cvt_f32_to_bf16<<<cvtBlocks, 256, 0, stream>>>(wo, wob);

  dim3 gg(DIM / 128, SEQ / 128);          // 16 x 16 blocks
  gemm_xwt<false><<<gg, 256, 0, stream>>>(xb, wqb, (void*)Qb);
  gemm_xwt<false><<<gg, 256, 0, stream>>>(xb, wkb, (void*)Kb);
  gemm_xwt<false><<<gg, 256, 0, stream>>>(xb, wvb, (void*)Vb);

  fa_kernel<<<NHEAD * (SEQ / 128), 256, 0, stream>>>(Qb, Kb, Vb, Ob);

  gemm_xwt<true><<<gg, 256, 0, stream>>>(Ob, wob, (void*)out);
}
